// Whitening2d_32066225831938
// MI455X (gfx1250) — matrix operand order repaired and checked by reading, not compiled, not run on hardware
//
#include <hip/hip_runtime.h>
#include <math.h>

// ---------------------------------------------------------------------------
// Whitening2d for MI455X (gfx1250, wave32, WMMA).
//   Big GEMMs: split-bf16 (hi+lo, 3 wmma products) -> v_wmma_f32_16x16x32_bf16
//     - fp32 -> bf16 hi/lo conversion done ONCE at LDS staging time
//     - LDS panels stored K-contiguous per column so fragments are raw
//       ds_load_b128, no per-fragment VALU conversion
//   Newton-Schulz 256x256 chain: native fp32       -> v_wmma_f32_16x16x4_f32
// ---------------------------------------------------------------------------

typedef __attribute__((ext_vector_type(16))) __bf16 v16bf;
typedef __attribute__((ext_vector_type(8)))  __bf16 v8bf;
typedef __attribute__((ext_vector_type(8)))  float  v8f;
typedef __attribute__((ext_vector_type(2)))  float  v2f;

constexpr int   D       = 256;        // feature dim
constexpr int   NROWS   = 262144;     // samples
constexpr int   KSTEP   = 32;         // K-panel (bf16 wmma K)
constexpr int   KP      = 40;         // padded K stride (bf16): 80 B, 16B-aligned
constexpr int   COVBLK  = 512;        // cov workgroups
constexpr int   RPB     = NROWS / COVBLK;  // rows per cov block (512)

// ----------------------------- WMMA helpers --------------------------------

__device__ __forceinline__ v8f wmma_bf16(v16bf a, v16bf b, v8f c) {
  return __builtin_amdgcn_wmma_f32_16x16x32_bf16(false, a, false, b,
                                                 (short)0, c, false, false);
}

__device__ __forceinline__ v16bf concat8(v8bf a, v8bf b) {
  return __builtin_shufflevector(a, b, 0, 1, 2, 3, 4, 5, 6, 7,
                                 8, 9, 10, 11, 12, 13, 14, 15);
}

// A fragment (16x32 bf16, M x K) from a [col][k] bf16 panel (K-stride KP).
// ISA A layout: lane half h, m = lane%16; VGPRs 0-3 hold K=8h..8h+7,
// VGPRs 4-7 hold K=16+8h..16+8h+7  -> two contiguous 16-byte runs.
__device__ __forceinline__ v16bf load_a_frag(const __bf16* panel, int col) {
  const int lane = threadIdx.x & 31;
  const int h = lane >> 4, m = lane & 15;
  const __bf16* p = panel + (col + m) * KP;
  v8bf r0 = *(const v8bf*)(p + 8 * h);
  v8bf r1 = *(const v8bf*)(p + 16 + 8 * h);
  return concat8(r0, r1);
}

// B fragment (32x16 bf16, K x N) from a [col][k] bf16 panel.
// B layout (matches A): lane half h, n=lane%16; elements 0-7 hold K=8h..8h+7,
// elements 8-15 hold K=16+8h..16+8h+7 -> two contiguous 16-byte runs.
__device__ __forceinline__ v16bf load_b_frag(const __bf16* panel, int col) {
  const int lane = threadIdx.x & 31;
  const int h = lane >> 4, n = lane & 15;
  const __bf16* p = panel + (col + n) * KP;
  v8bf r0 = *(const v8bf*)(p + 8 * h);
  v8bf r1 = *(const v8bf*)(p + 16 + 8 * h);
  return concat8(r0, r1);
}

__device__ __forceinline__ void split_bf16(float v, __bf16& hi, __bf16& lo) {
  hi = (__bf16)v;
  lo = (__bf16)(v - (float)hi);
}

// ----------------------------- tiny kernels --------------------------------

__global__ void k_zero(float* __restrict__ p, int n) {
  int i = blockIdx.x * blockDim.x + threadIdx.x;
  if (i < n) p[i] = 0.0f;
}

// sigma = (G - colsum_i*colsum_j/N) / (N-1)   (in place over G)
__global__ void k_finalize_sigma(float* __restrict__ G,
                                 const float* __restrict__ colsum) {
  int e = blockIdx.x * 256 + threadIdx.x;
  int i = e >> 8, j = e & 255;
  float corr = colsum[i] * colsum[j] * (1.0f / (float)NROWS);
  G[e] = (G[e] - corr) * (1.0f / (float)(NROWS - 1));
}

// scalars[0]=trace, [1]=1/trace, [2]=rsqrt(trace)
__global__ void k_trace(const float* __restrict__ sigma,
                        float* __restrict__ scalars) {
  __shared__ float red[256];
  int t = threadIdx.x;
  red[t] = sigma[t * (D + 1)];
  __syncthreads();
  for (int s = 128; s > 0; s >>= 1) {
    if (t < s) red[t] += red[t + s];
    __syncthreads();
  }
  if (t == 0) {
    float tr = red[0];
    scalars[0] = tr;
    scalars[1] = 1.0f / tr;
    scalars[2] = rsqrtf(tr);
  }
}

// Sn = sigma/trace ;  P = I
__global__ void k_prep(const float* __restrict__ sigma,
                       const float* __restrict__ scalars,
                       float* __restrict__ Sn, float* __restrict__ P) {
  int e = blockIdx.x * 256 + threadIdx.x;
  float it = scalars[1];
  Sn[e] = sigma[e] * it;
  P[e] = ((e >> 8) == (e & 255)) ? 1.0f : 0.0f;
}

// P = 1.5 P - 0.5 T
__global__ void k_update(float* __restrict__ P, const float* __restrict__ T) {
  int e = blockIdx.x * 256 + threadIdx.x;
  P[e] = 1.5f * P[e] - 0.5f * T[e];
}

// wm = P * rsqrt(trace)
__global__ void k_scale_wm(const float* __restrict__ P,
                           const float* __restrict__ scalars,
                           float* __restrict__ wm) {
  int e = blockIdx.x * 256 + threadIdx.x;
  wm[e] = P[e] * scalars[2];
}

// ------------------- Gram matrix: G += x^T x (split-bf16) -------------------
// grid COVBLK x 1024 threads (32 waves). Each block: RPB rows of x. A 32-row
// panel is staged per K-step, converted to bf16 hi/lo, stored TRANSPOSED as
// [col][k] so both A (= x^T tile) and B fragments are contiguous LDS loads.
// Wave w covers output tile-row (w&15), column half (w>>4): 8 16x16 tiles.
// Column sums fold into the staging loads (each thread owns one column).
__global__ void __launch_bounds__(1024)
k_cov(const float* __restrict__ x, float* __restrict__ G,
      float* __restrict__ colsum) {
  __shared__ __attribute__((aligned(16))) __bf16 pHi[D * KP];  // 20 KB
  __shared__ __attribute__((aligned(16))) __bf16 pLo[D * KP];  // 20 KB

  const int t    = threadIdx.x;
  const int wave = t >> 5;
  const int lane = t & 31;
  const int h    = lane >> 4;
  const int n    = lane & 15;
  const int M0   = (wave & 15) * 16;   // output tile row
  const int N0   = (wave >> 4) * 128;  // output column half
  const int scol = t & 255;            // staging column owned by this thread
  const int srow = t >> 8;             // staging row group (0..3)

  v8f acc[8];
#pragma unroll
  for (int i = 0; i < 8; ++i)
#pragma unroll
    for (int j = 0; j < 8; ++j) acc[i][j] = 0.0f;

  float cs = 0.0f;  // partial column sum for column scol
  const size_t base = (size_t)blockIdx.x * RPB;

  for (int k0 = 0; k0 < RPB; k0 += KSTEP) {
    __syncthreads();
    const float* src = x + (base + (size_t)k0) * D;
#pragma unroll
    for (int i = 0; i < 8; ++i) {
      const int row = srow + 4 * i;         // k within panel
      const float v = src[row * D + scol];  // coalesced across t
      cs += v;
      __bf16 hi, lo;
      split_bf16(v, hi, lo);
      pHi[scol * KP + row] = hi;            // transposed store
      pLo[scol * KP + row] = lo;
    }
    if (k0 + KSTEP < RPB)
      __builtin_prefetch(src + (size_t)KSTEP * D + t * 8, 0, 1);
    __syncthreads();

    const v16bf ahi = load_a_frag(pHi, M0);
    const v16bf alo = load_a_frag(pLo, M0);
#pragma unroll
    for (int tc = 0; tc < 8; ++tc) {
      const v16bf bhi = load_b_frag(pHi, N0 + tc * 16);
      const v16bf blo = load_b_frag(pLo, N0 + tc * 16);
      acc[tc] = wmma_bf16(ahi, bhi, acc[tc]);
      acc[tc] = wmma_bf16(ahi, blo, acc[tc]);
      acc[tc] = wmma_bf16(alo, bhi, acc[tc]);
    }
  }

  atomicAdd(&colsum[scol], cs);
#pragma unroll
  for (int tc = 0; tc < 8; ++tc)
#pragma unroll
    for (int r = 0; r < 8; ++r)
      atomicAdd(&G[(size_t)(M0 + r + 8 * h) * D + N0 + tc * 16 + n],
                acc[tc][r]);
}

// -------------------- 256x256x256 fp32 GEMM: C = A @ B ----------------------
// One wave (32 threads) per 16x16 output tile; grid (16,16).
// f32 WMMA 16x16x4: A lane(m=lane%16): V={K=2h, K=2h+1}; B: V={K=h, K=h+2}.
__global__ void __launch_bounds__(32)
k_gemm256(const float* __restrict__ A, const float* __restrict__ B,
          float* __restrict__ C) {
  const int lane = threadIdx.x;
  const int h = lane >> 4, q = lane & 15;
  const int bm = blockIdx.y, bn = blockIdx.x;

  v8f acc;
#pragma unroll
  for (int r = 0; r < 8; ++r) acc[r] = 0.0f;

  for (int k0 = 0; k0 < D; k0 += 4) {
    v2f a, b;
    a[0] = A[(bm * 16 + q) * D + k0 + 2 * h];
    a[1] = A[(bm * 16 + q) * D + k0 + 2 * h + 1];
    b[0] = B[(k0 + h) * D + bn * 16 + q];
    b[1] = B[(k0 + 2 + h) * D + bn * 16 + q];
    acc = __builtin_amdgcn_wmma_f32_16x16x4_f32(false, a, false, b,
                                                (short)0, acc, false, false);
  }
#pragma unroll
  for (int r = 0; r < 8; ++r)
    C[(bm * 16 + r + 8 * h) * D + bn * 16 + q] = acc[r];
}

// ------------------ out = (x - mean) @ wm  (split-bf16) ---------------------
// 4096 blocks x 256 threads (8 waves). Block covers 64 rows x 256 cols.
// A panel (centered x) stored [m][k] bf16 hi/lo; wm panel stored transposed
// [n][k] bf16 hi/lo. Wave w: tile-row w&3, column half w>>2 -> 8 tiles.
__global__ void __launch_bounds__(256)
k_out(const float* __restrict__ x, const float* __restrict__ wm,
      const float* __restrict__ colsum, float* __restrict__ out) {
  __shared__ __attribute__((aligned(16))) __bf16 aHi[64 * KP];  // 5 KB
  __shared__ __attribute__((aligned(16))) __bf16 aLo[64 * KP];  // 5 KB
  __shared__ __attribute__((aligned(16))) __bf16 bHi[D * KP];   // 20 KB
  __shared__ __attribute__((aligned(16))) __bf16 bLo[D * KP];   // 20 KB

  const int t    = threadIdx.x;
  const int wave = t >> 5;
  const int lane = t & 31;
  const int h    = lane >> 4;
  const int n    = lane & 15;
  const int M0   = (wave & 3) * 16;   // local tile row (of 64)
  const int N0   = (wave >> 2) * 128; // column half
  const size_t rowbase = (size_t)blockIdx.x * 64;
  const float invN = 1.0f / (float)NROWS;

  v8f acc[8];
#pragma unroll
  for (int i = 0; i < 8; ++i)
#pragma unroll
    for (int j = 0; j < 8; ++j) acc[i][j] = 0.0f;

  for (int k0 = 0; k0 < D; k0 += KSTEP) {
    __syncthreads();
    // stage centered A panel: [row m][k] bf16 hi/lo
#pragma unroll
    for (int i = 0; i < 8; ++i) {
      const int lin = t + i * 256;
      const int r = lin >> 5, c = lin & 31;
      const int f = k0 + c;
      const float v = x[(rowbase + r) * D + f] - colsum[f] * invN;
      __bf16 hi, lo;
      split_bf16(v, hi, lo);
      aHi[r * KP + c] = hi;
      aLo[r * KP + c] = lo;
    }
    // stage wm panel transposed: [col n][k] bf16 hi/lo
#pragma unroll
    for (int i = 0; i < 32; ++i) {
      const int lin = t + i * 256;
      const int k = lin >> 8, c = lin & 255;
      const float v = wm[(size_t)k0 * D + lin];
      __bf16 hi, lo;
      split_bf16(v, hi, lo);
      bHi[c * KP + k] = hi;
      bLo[c * KP + k] = lo;
    }
    __syncthreads();

    const v16bf ahi = load_a_frag(aHi, M0);
    const v16bf alo = load_a_frag(aLo, M0);
#pragma unroll
    for (int tc = 0; tc < 8; ++tc) {
      const v16bf bhi = load_b_frag(bHi, N0 + tc * 16);
      const v16bf blo = load_b_frag(bLo, N0 + tc * 16);
      acc[tc] = wmma_bf16(ahi, bhi, acc[tc]);
      acc[tc] = wmma_bf16(ahi, blo, acc[tc]);
      acc[tc] = wmma_bf16(alo, bhi, acc[tc]);
    }
  }

#pragma unroll
  for (int tc = 0; tc < 8; ++tc)
#pragma unroll
    for (int r = 0; r < 8; ++r)
      out[(rowbase + M0 + r + 8 * h) * D + N0 + tc * 16 + n] = acc[tc][r];
}

// ------------------------------- launcher -----------------------------------

extern "C" void kernel_launch(void* const* d_in, const int* in_sizes, int n_in,
                              void* d_out, int out_size, void* d_ws,
                              size_t ws_size, hipStream_t stream) {
  (void)in_sizes; (void)n_in; (void)out_size; (void)ws_size;

  const float* x = (const float*)d_in[0];
  float* out = (float*)d_out;

  char* ws = (char*)d_ws;                     // layout (bytes):
  float* colsum  = (float*)(ws + 0);          //   256 f
  float* scalars = (float*)(ws + 2048);       //   trace, 1/t, rsqrt(t)
  float* G       = (float*)(ws + 4096);       //   Gram -> sigma (in place)
  float* Sn      = G  + 65536;                //   sigma / trace
  float* P       = Sn + 65536;                //   NS iterate
  float* Ta      = P  + 65536;                //   temp / wm
  float* Tb      = Ta + 65536;                //   temp

  // zero colsum + scalars + Gram accumulator (atomics targets)
  {
    const int nz = 1024 + 65536;  // first 4 KB (as floats) + G
    k_zero<<<(nz + 255) / 256, 256, 0, stream>>>((float*)ws, nz);
  }

  // Gram matrix + column sums in one pass over x
  k_cov<<<COVBLK, 1024, 0, stream>>>(x, G, colsum);

  // sigma, trace, sigma_n, P = I
  k_finalize_sigma<<<256, 256, 0, stream>>>(G, colsum);
  k_trace<<<1, 256, 0, stream>>>(G, scalars);
  k_prep<<<256, 256, 0, stream>>>(G, scalars, Sn, P);

  // 4 Newton-Schulz iterations: P = 1.5 P - 0.5 (P @ P @ P @ Sn)
  dim3 g16(16, 16);
  for (int it = 0; it < 4; ++it) {
    k_gemm256<<<g16, 32, 0, stream>>>(P, P, Ta);    // Ta = P P
    k_gemm256<<<g16, 32, 0, stream>>>(Ta, P, Tb);   // Tb = P P P
    k_gemm256<<<g16, 32, 0, stream>>>(Tb, Sn, Ta);  // Ta = P^3 Sn
    k_update<<<256, 256, 0, stream>>>(P, Ta);
  }

  // wm = P * rsqrt(trace)  (into Ta), then out = (x - mean) @ wm
  k_scale_wm<<<256, 256, 0, stream>>>(P, scalars, Ta);
  k_out<<<NROWS / 64, 256, 0, stream>>>(x, Ta, colsum, out);
}
